// CosineSimilarity2D_7567732376287
// MI455X (gfx1250) — matrix operand order repaired and checked by reading, not compiled, not run on hardware
//
#include <hip/hip_runtime.h>
#include <hip/hip_bf16.h>
#include <stdint.h>

// ---------------------------------------------------------------------------
// CosineSimilarity2D on MI455X (gfx1250): conv-as-GEMM, bf16 split-precision
// (hi+lo) WMMA: 3x v_wmma_f32_16x16x32_bf16 per fp32 K=32 slice.
// Fast path: x pre-split into bf16 hi/lo arrays (padded-S layout makes every
// A-fragment run 8 contiguous bf16 -> 1x global_load_b128), sliding-window
// ||x|| computed separably outside the loop. Hot loop = 4 A loads + 4 B loads
// + 6 WMMA + uniform SALU. Fallback (small ws): round-2 fused kernel.
// ---------------------------------------------------------------------------

typedef __attribute__((ext_vector_type(16))) __bf16   v16bf;
typedef __attribute__((ext_vector_type(8)))  float    v8f;
typedef __attribute__((ext_vector_type(8)))  uint32_t v8u;
typedef __attribute__((ext_vector_type(4)))  uint32_t u4v;
typedef __attribute__((ext_vector_type(4)))  float    f4a;

#define B_     32
#define C_     64
#define H_     96
#define W_     96
#define K_     32            // prototypes
#define R_     13
#define S_     13
#define SP_    16            // padded kernel width (s = 13..15 are zero in B)
#define OH_    84
#define OW_    84
#define RED_   (C_*R_*S_)        // 10816 true reduction length
#define GRP_   (C_*R_)           // 832 (c,r) groups
#define REDP_  (GRP_*SP_)        // 13312 padded reduction length
#define NCHUNK (REDP_/32)        // 416 K-chunks of 32 (2 groups each)
#define IMG_   (C_*H_*W_)        // 589824 floats per image
#define XTOT_  (B_*IMG_)         // 18874368 floats total
#define MTOT   (B_*OH_*OW_)      // 225792 output positions
#define MTILES (MTOT/16)         // 14112 (exact)
#define FRAG_DWORDS (2*NCHUNK*32*8)  // 212992 dwords per precision part

// ws layout (bytes)
#define OFF_INVN   0u
#define OFF_INVXN  256u                                   // MTOT floats
#define OFF_BHI    (OFF_INVXN + MTOT*4u)                  // 903424
#define OFF_BLO    (OFF_BHI + FRAG_DWORDS*4u)
#define OFF_Z      (OFF_BLO + FRAG_DWORDS*4u)             // B*H*W floats
#define OFF_HZ     (OFF_Z + B_*H_*W_*4u)                  // B*H*OW floats
#define OFF_XHI    (OFF_HZ + B_*H_*OW_*4u)                // XTOT u16
#define OFF_XLO    (OFF_XHI + XTOT_*2u)                   // XTOT u16
#define WS_FAST    ((size_t)OFF_XLO + XTOT_*2u + 64u)     // ~80.4 MB
#define WS_FALLBACK ((size_t)OFF_BLO + FRAG_DWORDS*4u)    // ~2.6 MB

// Truncation split: v = bf16(hi) + ~bf16(lo); hi = top 16 bits of fp32.
__device__ __forceinline__ void split_hi_lo(float v, uint32_t& h, uint32_t& l) {
    uint32_t u = __float_as_uint(v);
    h = u >> 16;
    float lf = v - __uint_as_float(u & 0xFFFF0000u);
    l = __float_as_uint(lf) >> 16;
}

// ---- per-prototype 1/max(||y||, EPS) ----------------------------------------
__global__ void proto_norms(const float* __restrict__ y, float* __restrict__ invn) {
    __shared__ float red[256];
    int p = blockIdx.x;
    const float* yp = y + (size_t)p * RED_;
    float s = 0.f;
    for (int i = threadIdx.x; i < RED_; i += 256) { float v = yp[i]; s += v * v; }
    red[threadIdx.x] = s;
    __syncthreads();
    for (int off = 128; off > 0; off >>= 1) {
        if (threadIdx.x < off) red[threadIdx.x] += red[threadIdx.x + off];
        __syncthreads();
    }
    if (threadIdx.x == 0) invn[p] = 1.0f / fmaxf(sqrtf(red[0]), 1e-9f);
}

// ---- pack normed_y into WMMA B fragments (hi/lo bf16, S padded to 16) --------
// B (32x16 bf16): lanes 0-15 = cols, K=0..7,16..23; lanes 16-31 = same cols, K=8..15,24..31;
// VGPR v packs K=16*(v>>2)+8*half+2*(v&3) (lo16), +1 (hi16).  dword idx = ((chunk*2+nt)*32+lane)*8+v
__global__ void pack_b(const float* __restrict__ y, const float* __restrict__ invn,
                       uint32_t* __restrict__ bhi, uint32_t* __restrict__ blo) {
    int idx   = blockIdx.x * 256 + threadIdx.x;      // < FRAG_DWORDS
    int v     = idx & 7;
    int lane  = (idx >> 3) & 31;
    int nt    = (idx >> 8) & 1;
    int chunk = idx >> 9;
    int n     = nt * 16 + (lane & 15);
    int khalf = (lane >= 16) ? 8 : 0;
    int kp0   = chunk * 32 + (v >> 2) * 16 + khalf + 2 * (v & 3);  // padded reduction index
    float sc  = invn[n];
    float vals[2];
    #pragma unroll
    for (int e = 0; e < 2; ++e) {
        int kp = kp0 + e;
        int g  = kp >> 4;            // (c,r) group
        int s  = kp & 15;            // padded s
        int c  = g / R_;
        int r  = g - c * R_;
        vals[e] = (s < S_) ? y[(size_t)n * RED_ + c * (R_*S_) + r * S_ + s] * sc : 0.0f;
    }
    uint32_t h0, l0, h1, l1;
    split_hi_lo(vals[0], h0, l0);
    split_hi_lo(vals[1], h1, l1);
    bhi[idx] = h0 | (h1 << 16);
    blo[idx] = l0 | (l1 << 16);
}

// ---- fast-path prep: split x into bf16 hi/lo arrays (4 floats/thread) --------
__global__ void xsplit(const float* __restrict__ x, uint32_t* __restrict__ xhi,
                       uint32_t* __restrict__ xlo) {
    int t = blockIdx.x * 256 + threadIdx.x;          // < XTOT_/4
    f4a q = *(const f4a*)(x + (size_t)t * 4);
    uint32_t h[4], l[4];
    split_hi_lo(q.x, h[0], l[0]);
    split_hi_lo(q.y, h[1], l[1]);
    split_hi_lo(q.z, h[2], l[2]);
    split_hi_lo(q.w, h[3], l[3]);
    xhi[2*t]   = h[0] | (h[1] << 16);
    xhi[2*t+1] = h[2] | (h[3] << 16);
    xlo[2*t]   = l[0] | (l[1] << 16);
    xlo[2*t+1] = l[2] | (l[3] << 16);
}

// ---- fast-path prep: separable sliding-window ||x|| --------------------------
__global__ void csq(const float* __restrict__ x, float* __restrict__ z) {
    int t = blockIdx.x * 256 + threadIdx.x;          // < B*H*W
    int b = t / (H_ * W_);
    int hw = t - b * (H_ * W_);
    const float* xp = x + (size_t)b * IMG_ + hw;
    float s = 0.f;
    #pragma unroll 8
    for (int c = 0; c < C_; ++c) { float v = xp[c * (H_ * W_)]; s = fmaf(v, v, s); }
    z[t] = s;
}
__global__ void hsum(const float* __restrict__ z, float* __restrict__ hz) {
    int t = blockIdx.x * 256 + threadIdx.x;          // < B*H*OW
    int b = t / (H_ * OW_);
    int rem = t - b * (H_ * OW_);
    int h = rem / OW_;
    int w = rem - h * OW_;
    const float* zp = z + b * (H_ * W_) + h * W_ + w;
    float s = 0.f;
    #pragma unroll
    for (int i = 0; i < S_; ++i) s += zp[i];
    hz[t] = s;
}
__global__ void vnorm(const float* __restrict__ hz, float* __restrict__ invxn) {
    int t = blockIdx.x * 256 + threadIdx.x;          // < MTOT
    int b = t / (OH_ * OW_);
    int rem = t - b * (OH_ * OW_);
    int oh = rem / OW_;
    int ow = rem - oh * OW_;
    const float* hp = hz + b * (H_ * OW_) + oh * OW_ + ow;
    float s = 0.f;
    #pragma unroll
    for (int i = 0; i < R_; ++i) s += hp[i * OW_];
    invxn[t] = 1.0f / fmaxf(sqrtf(s), 1e-9f);
}

// ---- fast-path main: pure WMMA + b128 loads ---------------------------------
__global__ void __launch_bounds__(256)
cosim_main_ps(const uint16_t* __restrict__ xhi, const uint16_t* __restrict__ xlo,
              const uint32_t* __restrict__ bhi, const uint32_t* __restrict__ blo,
              const float* __restrict__ invxn, float* __restrict__ out) {
    const int lane  = threadIdx.x & 31;
    const int mtile = blockIdx.x * 8 + (threadIdx.x >> 5);

    const int mrow = lane & 15;
    const int m    = mtile * 16 + mrow;
    const int b    = m / (OH_ * OW_);
    const int pos  = m - b * (OH_ * OW_);
    const int oh   = pos / OW_;
    const int ow   = pos - oh * OW_;
    const bool upper = lane >= 16;                   // upper lanes supply K=8..15
    // element index of this lane's run start within the split arrays
    const int posoff = b * IMG_ + oh * W_ + ow + (upper ? 8 : 0);

    v8f acc0 = {}, acc1 = {};

    for (int chunk = 0; chunk < NCHUNK; ++chunk) {
        v8u ah_u, al_u;
        #pragma unroll
        for (int j = 0; j < 2; ++j) {
            const int g    = chunk * 2 + j;          // uniform -> SALU
            const int c    = g / R_;
            const int r    = g - c * R_;
            const int bs   = posoff + c * (H_ * W_) + r * W_;
            // 8 contiguous bf16 = one b128 per precision part per run.
            // Upper lanes' s=13..15 slots may read garbage (zero B rows).
            u4v qh = *(const u4v*)(xhi + bs);
            u4v ql = *(const u4v*)(xlo + bs);
            #pragma unroll
            for (int i = 0; i < 4; ++i) { ah_u[4*j+i] = qh[i]; al_u[4*j+i] = ql[i]; }
        }
        v16bf a_hi = __builtin_bit_cast(v16bf, ah_u);
        v16bf a_lo = __builtin_bit_cast(v16bf, al_u);

        const uint32_t fbase = ((uint32_t)chunk * 2) * 256 + (uint32_t)lane * 8;
        v16bf b_hi0 = __builtin_bit_cast(v16bf, *(const v8u*)(bhi + fbase));
        v16bf b_hi1 = __builtin_bit_cast(v16bf, *(const v8u*)(bhi + fbase + 256));
        v16bf b_lo0 = __builtin_bit_cast(v16bf, *(const v8u*)(blo + fbase));
        v16bf b_lo1 = __builtin_bit_cast(v16bf, *(const v8u*)(blo + fbase + 256));

        acc0 = __builtin_amdgcn_wmma_f32_16x16x32_bf16(false, a_hi, false, b_hi0, (short)0, acc0, false, false);
        acc0 = __builtin_amdgcn_wmma_f32_16x16x32_bf16(false, a_hi, false, b_lo0, (short)0, acc0, false, false);
        acc0 = __builtin_amdgcn_wmma_f32_16x16x32_bf16(false, a_lo, false, b_hi0, (short)0, acc0, false, false);
        acc1 = __builtin_amdgcn_wmma_f32_16x16x32_bf16(false, a_hi, false, b_hi1, (short)0, acc1, false, false);
        acc1 = __builtin_amdgcn_wmma_f32_16x16x32_bf16(false, a_hi, false, b_lo1, (short)0, acc1, false, false);
        acc1 = __builtin_amdgcn_wmma_f32_16x16x32_bf16(false, a_lo, false, b_hi1, (short)0, acc1, false, false);
    }

    const int nc    = lane & 15;
    const int rbase = upper ? 8 : 0;
    #pragma unroll
    for (int v = 0; v < 8; ++v) {
        const int row  = rbase + v;
        const int mg   = mtile * 16 + row;
        const float invr = invxn[mg];
        const int bb   = mg / (OH_ * OW_);
        const int pp   = mg - bb * (OH_ * OW_);
        const int ohh  = pp / OW_;
        const int oww  = pp - ohh * OW_;
        const float d0 = fmaxf(acc0[v] * invr, 0.0f);
        const float d1 = fmaxf(acc1[v] * invr, 0.0f);
        const size_t o0 = (((size_t)bb * K_ + nc)      * OH_ + ohh) * OW_ + oww;
        const size_t o1 = (((size_t)bb * K_ + nc + 16) * OH_ + ohh) * OW_ + oww;
        out[o0] = d0;
        out[o1] = d1;
    }
}

// ---- fallback main (round-2 fused version, small ws) -------------------------
__global__ void __launch_bounds__(256)
cosim_main_fused(const float* __restrict__ x,
                 const uint32_t* __restrict__ bhi, const uint32_t* __restrict__ blo,
                 float* __restrict__ out) {
    const int lane  = threadIdx.x & 31;
    const int mtile = blockIdx.x * 8 + (threadIdx.x >> 5);
    const int mrow = lane & 15;
    const int m    = mtile * 16 + mrow;
    const int b    = m / (OH_ * OW_);
    const int pos  = m - b * (OH_ * OW_);
    const int oh   = pos / OW_;
    const int ow   = pos - oh * OW_;
    const float* xb = x + (size_t)b * IMG_;
    const bool  upper  = lane >= 16;
    const int   posoff = oh * W_ + ow + (upper ? 8 : 0);
    const int   rlim   = (B_ - b) * IMG_ - 4;

    v8f acc0 = {}, acc1 = {};
    float ssqa = 0.f, ssqt = 0.f;

    for (int chunk = 0; chunk < NCHUNK; ++chunk) {
        float e[16];
        int base1;
        #pragma unroll
        for (int j = 0; j < 2; ++j) {
            const int g    = chunk * 2 + j;
            const int c    = g / R_;
            const int r    = g - c * R_;
            const int bs   = posoff + c * (H_ * W_) + r * W_;
            if (j == 1) base1 = bs;
            const int bs2  = min(bs + 4, rlim);
            f4a qa = *(const f4a*)(xb + bs);
            f4a qb = *(const f4a*)(xb + bs2);
            e[8*j+0] = qa.x; e[8*j+1] = qa.y; e[8*j+2] = qa.z; e[8*j+3] = qa.w;
            e[8*j+4] = qb.x; e[8*j+5] = qb.y; e[8*j+6] = qb.z; e[8*j+7] = qb.w;
        }
        if (chunk == NCHUNK - 1) {
            float s12 = xb[base1 + 4];
            e[12] = upper ? s12 : e[12];
        }
        v8u ah_u, al_u;
        #pragma unroll
        for (int p = 0; p < 8; ++p) {
            float e0 = e[2*p], e1 = e[2*p+1];
            if ((2*p & 7) <= 4)  ssqa = fmaf(e0, e0, ssqa); else ssqt = fmaf(e0, e0, ssqt);
            if (((2*p+1) & 7) <= 4) ssqa = fmaf(e1, e1, ssqa); else ssqt = fmaf(e1, e1, ssqt);
            uint32_t u0 = __float_as_uint(e0), u1 = __float_as_uint(e1);
            ah_u[p] = __builtin_amdgcn_perm(u1, u0, 0x07060302u);
            float l0 = e0 - __uint_as_float(u0 & 0xFFFF0000u);
            float l1 = e1 - __uint_as_float(u1 & 0xFFFF0000u);
            al_u[p] = __builtin_amdgcn_perm(__float_as_uint(l1), __float_as_uint(l0), 0x07060302u);
        }
        v16bf a_hi = __builtin_bit_cast(v16bf, ah_u);
        v16bf a_lo = __builtin_bit_cast(v16bf, al_u);
        const uint32_t fbase = ((uint32_t)chunk * 2) * 256 + (uint32_t)lane * 8;
        v16bf b_hi0 = __builtin_bit_cast(v16bf, *(const v8u*)(bhi + fbase));
        v16bf b_hi1 = __builtin_bit_cast(v16bf, *(const v8u*)(bhi + fbase + 256));
        v16bf b_lo0 = __builtin_bit_cast(v16bf, *(const v8u*)(blo + fbase));
        v16bf b_lo1 = __builtin_bit_cast(v16bf, *(const v8u*)(blo + fbase + 256));
        acc0 = __builtin_amdgcn_wmma_f32_16x16x32_bf16(false, a_hi, false, b_hi0, (short)0, acc0, false, false);
        acc0 = __builtin_amdgcn_wmma_f32_16x16x32_bf16(false, a_hi, false, b_lo0, (short)0, acc0, false, false);
        acc0 = __builtin_amdgcn_wmma_f32_16x16x32_bf16(false, a_lo, false, b_hi0, (short)0, acc0, false, false);
        acc1 = __builtin_amdgcn_wmma_f32_16x16x32_bf16(false, a_hi, false, b_hi1, (short)0, acc1, false, false);
        acc1 = __builtin_amdgcn_wmma_f32_16x16x32_bf16(false, a_hi, false, b_lo1, (short)0, acc1, false, false);
        acc1 = __builtin_amdgcn_wmma_f32_16x16x32_bf16(false, a_lo, false, b_hi1, (short)0, acc1, false, false);
    }

    float ssq = ssqa + (upper ? 0.f : ssqt);
    ssq += __shfl_xor(ssq, 16, 32);
    const float inv = 1.0f / fmaxf(sqrtf(ssq), 1e-9f);

    const int nc    = lane & 15;
    const int rbase = upper ? 8 : 0;
    #pragma unroll
    for (int v = 0; v < 8; ++v) {
        const int row  = rbase + v;
        const float invr = __shfl(inv, row, 32);
        const int mg   = mtile * 16 + row;
        const int bb   = mg / (OH_ * OW_);
        const int pp   = mg - bb * (OH_ * OW_);
        const int ohh  = pp / OW_;
        const int oww  = pp - ohh * OW_;
        const float d0 = fmaxf(acc0[v] * invr, 0.0f);
        const float d1 = fmaxf(acc1[v] * invr, 0.0f);
        const size_t o0 = (((size_t)bb * K_ + nc)      * OH_ + ohh) * OW_ + oww;
        const size_t o1 = (((size_t)bb * K_ + nc + 16) * OH_ + ohh) * OW_ + oww;
        out[o0] = d0;
        out[o1] = d1;
    }
}

// ---------------------------------------------------------------------------
extern "C" void kernel_launch(void* const* d_in, const int* in_sizes, int n_in,
                              void* d_out, int out_size, void* d_ws, size_t ws_size,
                              hipStream_t stream) {
    const float* x = (const float*)d_in[0];
    const float* y = (const float*)d_in[1];
    float* out = (float*)d_out;

    char* ws = (char*)d_ws;
    float*    invn  = (float*)(ws + OFF_INVN);
    float*    invxn = (float*)(ws + OFF_INVXN);
    uint32_t* bhi   = (uint32_t*)(ws + OFF_BHI);
    uint32_t* blo   = (uint32_t*)(ws + OFF_BLO);
    float*    z     = (float*)(ws + OFF_Z);
    float*    hz    = (float*)(ws + OFF_HZ);
    uint32_t* xhi   = (uint32_t*)(ws + OFF_XHI);
    uint32_t* xlo   = (uint32_t*)(ws + OFF_XLO);

    proto_norms<<<K_, 256, 0, stream>>>(y, invn);
    pack_b<<<FRAG_DWORDS / 256, 256, 0, stream>>>(y, invn, bhi, blo);

    if (ws_size >= WS_FAST) {
        xsplit<<<XTOT_ / 4 / 256, 256, 0, stream>>>(x, xhi, xlo);
        csq<<<(B_*H_*W_) / 256, 256, 0, stream>>>(x, z);
        hsum<<<(B_*H_*OW_) / 256, 256, 0, stream>>>(z, hz);
        vnorm<<<MTOT / 256, 256, 0, stream>>>(hz, invxn);
        cosim_main_ps<<<MTILES / 8, 256, 0, stream>>>(
            (const uint16_t*)xhi, (const uint16_t*)xlo, bhi, blo, invxn, out);
    } else {
        cosim_main_fused<<<MTILES / 8, 256, 0, stream>>>(x, bhi, blo, out);
    }
}
